// RSSM_7095285973494
// MI455X (gfx1250) — matrix operand order repaired and checked by reading, not compiled, not run on hardware
//
#include <hip/hip_runtime.h>
#include <math.h>

// ---------------------------------------------------------------- constants
#define Bsz    512
#define Tlen   64
#define STOCH  32
#define DETER  512
#define HIDDEN 512
#define ACTD   12
#define EMBED  1024
#define GRU3   1536
#define MB     16       // batch rows per workgroup (one WMMA M tile)
#define THREADS 512     // 16 waves (wave32) per workgroup
#define NW     16       // waves per workgroup
#define NWG    (Bsz / MB)

typedef __bf16 bf16;
typedef __attribute__((ext_vector_type(16))) __bf16 v16bf;
typedef __attribute__((ext_vector_type(8)))  __bf16 v8bf;
typedef __attribute__((ext_vector_type(8)))  float  v8f;

// workspace carve-out (bytes): bf16 transposed weights [N][Kp], then f32 init state
#define OFF_WIN    0u
#define OFF_WGRU   (OFF_WIN    + 512u * 64u   * 2u)
#define OFF_WOUT   (OFF_WGRU   + 1536u * 1024u * 2u)
#define OFF_WIMS   (OFF_WOUT   + 512u * 512u  * 2u)
#define OFF_WOBS   (OFF_WIMS   + 64u  * 512u  * 2u)
#define OFF_WOSTAT (OFF_WOBS   + 512u * 1536u * 2u)
#define OFF_DETER0 (OFF_WOSTAT + 64u  * 512u  * 2u)
#define OFF_STOCH0 (OFF_DETER0 + 512u * 4u)

__device__ __forceinline__ float sigmoidf_(float x) { return 1.f / (1.f + expf(-x)); }
__device__ __forceinline__ float softplusf_(float x) { return x > 20.f ? x : log1pf(expf(x)); }

// -------------------------------------------------- weight transpose+convert
// dst[n*Kp + k] = (k < K) ? bf16(w[k*N + n]) : 0   (pads K up to Kp with zeros)
__global__ void k_wt(const float* __restrict__ w, bf16* __restrict__ dst,
                     int K, int N, int Kp) {
  int idx = blockIdx.x * 256 + threadIdx.x;
  if (idx >= N * Kp) return;
  int n = idx / Kp, k = idx - n * Kp;
  dst[idx] = (k < K) ? (bf16)w[k * N + n] : (bf16)0.f;
}

// ------------------------------------------- initial state (deter0 / stoch0)
__global__ void k_init(const float* __restrict__ W_init,
                       const float* __restrict__ w_out,
                       const float* __restrict__ ln_out_g,
                       const float* __restrict__ ln_out_b,
                       const float* __restrict__ w_ims,
                       const float* __restrict__ b_ims,
                       float* __restrict__ deter0, float* __restrict__ stoch0) {
  __shared__ float sd[DETER], sx[DETER], red[256];
  const int tid = threadIdx.x;
  for (int i = tid; i < DETER; i += 256) { float d = tanhf(W_init[i]); sd[i] = d; deter0[i] = d; }
  __syncthreads();
  for (int n = tid; n < DETER; n += 256) {
    float acc = 0.f;
    for (int k = 0; k < DETER; ++k) acc += sd[k] * w_out[k * HIDDEN + n];
    sx[n] = acc;
  }
  __syncthreads();
  float p = 0.f, q = 0.f;
  for (int i = tid; i < DETER; i += 256) { float v = sx[i]; p += v; q += v * v; }
  red[tid] = p; __syncthreads();
  for (int s = 128; s > 0; s >>= 1) { if (tid < s) red[tid] += red[tid + s]; __syncthreads(); }
  float mean = red[0] / (float)HIDDEN; __syncthreads();
  red[tid] = q; __syncthreads();
  for (int s = 128; s > 0; s >>= 1) { if (tid < s) red[tid] += red[tid + s]; __syncthreads(); }
  float var = red[0] / (float)HIDDEN - mean * mean;
  float rstd = rsqrtf(var + 1e-3f);
  __syncthreads();
  for (int n = tid; n < HIDDEN; n += 256) {
    float y = (sx[n] - mean) * rstd * ln_out_g[n] + ln_out_b[n];
    sx[n] = y * sigmoidf_(y);
  }
  __syncthreads();
  if (tid < STOCH) {
    float acc = b_ims[tid];
    for (int k = 0; k < HIDDEN; ++k) acc += sx[k] * w_ims[k * (2 * STOCH) + tid];
    stoch0[tid] = acc;   // mean part only
  }
}

// ---------------------------------------------------------- WMMA tile GEMM
// OUT[16][N] = ACT[16][Kp](bf16,LDS) @ WT[N][Kp](bf16,global);  N-tiles split over 16 waves
__device__ __forceinline__ void wmma_gemm(const bf16* __restrict__ wT,
                                          const bf16* s_act, float* s_out,
                                          int outStride, int N, int Kp,
                                          int wave, int lane) {
  const int half = lane >> 4;
  const int l15  = lane & 15;
  const bf16* abase = s_act + l15 * Kp + half * 16;
  const int ntiles = N >> 4;
  for (int nt = wave; nt < ntiles; nt += NW) {
    const int n0 = nt << 4;
    const bf16* bbase = wT + (size_t)(n0 + l15) * Kp + half * 16;
    v8f acc = {};
    for (int k0 = 0; k0 < Kp; k0 += 32) {
      if ((k0 & 63) == 0) __builtin_prefetch(bbase + k0 + 512, 0, 0);
      union { v16bf v; v8bf h[2]; } au;
      au.h[0] = *(const v8bf*)(abase + k0);        // lanes0-15: K0-7 | lanes16-31: K16-23
      au.h[1] = *(const v8bf*)(abase + k0 + 8);    // lanes0-15: K8-15 | lanes16-31: K24-31
      v16bf bv = *(const v16bf*)(bbase + k0);      // lanes0-15: K0-15 | lanes16-31: K16-31
      acc = __builtin_amdgcn_wmma_f32_16x16x32_bf16(
          false, au.v, false, bv, (short)0, acc, false, false);
    }
    float* orow = s_out + half * 8 * outStride + (n0 + l15);
#pragma unroll
    for (int v = 0; v < 8; ++v) orow[v * outStride] = acc[v];  // row = half*8+v
  }
}

// row-wise mean / rstd over [16][N] f32 in LDS; one wave per row, shfl butterflies
__device__ __forceinline__ void ln_rows(const float* s_in, int N,
                                        float* s_mv, int tid) {
  const int r = tid >> 5;       // wave id == row
  const int j = tid & 31;
  float sum = 0.f, sq = 0.f;
  const float* row = s_in + r * N;
  for (int i = j; i < N; i += 32) { float v = row[i]; sum += v; sq += v * v; }
#pragma unroll
  for (int off = 16; off > 0; off >>= 1) {
    sum += __shfl_xor(sum, off, 32);
    sq  += __shfl_xor(sq,  off, 32);
  }
  if (j == 0) {
    float m = sum / (float)N, var = sq / (float)N - m * m;
    s_mv[r * 2] = m; s_mv[r * 2 + 1] = rsqrtf(var + 1e-3f);
  }
  __syncthreads();
}

// ------------------------------------------------------------ main persistent
__global__ __launch_bounds__(THREADS, 1) void k_rssm(
    const float* __restrict__ embed, const float* __restrict__ action,
    const float* __restrict__ is_first, const float* __restrict__ eps_prior,
    const float* __restrict__ eps_post,
    const float* __restrict__ ln_in_g, const float* __restrict__ ln_in_b,
    const float* __restrict__ ln_gru_g, const float* __restrict__ ln_gru_b,
    const float* __restrict__ ln_out_g, const float* __restrict__ ln_out_b,
    const float* __restrict__ b_ims,
    const float* __restrict__ ln_obs_g, const float* __restrict__ ln_obs_b,
    const float* __restrict__ b_ostat,
    const bf16* __restrict__ wT_in, const bf16* __restrict__ wT_gru,
    const bf16* __restrict__ wT_out, const bf16* __restrict__ wT_ims,
    const bf16* __restrict__ wT_obs, const bf16* __restrict__ wT_ostat,
    const float* __restrict__ deter0, const float* __restrict__ stoch0,
    float* __restrict__ out) {
  __shared__ float s_deter[MB * DETER];                      // 32 KB (carry / deter_n)
  __shared__ float s_stoch[MB * STOCH];                      // 2 KB
  __shared__ __align__(16) bf16 s_cat[MB * GRU3];            // 48 KB  bf16 A staging
  __shared__ float s_gemm[MB * GRU3];                        // 96 KB  f32 GEMM out
  __shared__ float s_stat[MB * 64];                          // 4 KB
  __shared__ float s_mv[16 * 2];
  __shared__ float s_f[16];

  const int tid = threadIdx.x;
  const int lane = tid & 31, wave = tid >> 5;
  const int b0 = blockIdx.x * MB;

  for (int i = tid; i < MB * DETER; i += THREADS) s_deter[i] = deter0[i & 511];
  for (int i = tid; i < MB * STOCH; i += THREADS) s_stoch[i] = stoch0[i & 31];
  __syncthreads();

  for (int t = 0; t < Tlen; ++t) {
    // ---- phase 0: is_first mix + build cat1 = [stoch, a*(1-f), pad] (K=64)
    if (tid < 16) s_f[tid] = is_first[(size_t)(b0 + tid) * Tlen + t];
    __syncthreads();
    for (int i = tid; i < MB * DETER; i += THREADS) {
      int r = i >> 9, c = i & 511; float f = s_f[r];
      s_deter[i] = s_deter[i] * (1.f - f) + deter0[c] * f;
    }
    for (int i = tid; i < MB * 64; i += THREADS) {
      int r = i >> 6, c = i & 63; float f = s_f[r]; float v;
      if (c < STOCH) { float s = s_stoch[r * STOCH + c] * (1.f - f) + stoch0[c] * f;
                       s_stoch[r * STOCH + c] = s; v = s; }
      else if (c < STOCH + ACTD)
        v = action[((size_t)(b0 + r) * Tlen + t) * ACTD + (c - STOCH)] * (1.f - f);
      else v = 0.f;
      s_cat[i] = (bf16)v;
    }
    __syncthreads();
    // ---- GEMM1: x = cat1 @ w_in  (N=512, K=64)
    wmma_gemm(wT_in, s_cat, s_gemm, HIDDEN, HIDDEN, 64, wave, lane);
    __syncthreads();
    // ---- LN(in)+silu -> cat2[:,0:512]; cat2[:,512:1024] = bf16(deter)
    ln_rows(s_gemm, HIDDEN, s_mv, tid);
    for (int i = tid; i < MB * HIDDEN; i += THREADS) {
      int r = i >> 9, c = i & 511;
      float y = (s_gemm[i] - s_mv[r * 2]) * s_mv[r * 2 + 1] * ln_in_g[c] + ln_in_b[c];
      y = y * sigmoidf_(y);
      s_cat[r * 1024 + c] = (bf16)y;
    }
    for (int i = tid; i < MB * DETER; i += THREADS) {
      int r = i >> 9, c = i & 511;
      s_cat[r * 1024 + 512 + c] = (bf16)s_deter[i];
    }
    __syncthreads();
    // ---- GEMM2: parts = cat2 @ w_gru (N=1536, K=1024)
    wmma_gemm(wT_gru, s_cat, s_gemm, GRU3, GRU3, 1024, wave, lane);
    __syncthreads();
    // ---- LN(gru) in-place
    ln_rows(s_gemm, GRU3, s_mv, tid);
    for (int i = tid; i < MB * GRU3; i += THREADS) {
      int r = i / GRU3, c = i - r * GRU3;
      s_gemm[i] = (s_gemm[i] - s_mv[r * 2]) * s_mv[r * 2 + 1] * ln_gru_g[c] + ln_gru_b[c];
    }
    __syncthreads();
    // ---- GRU gates -> deter_n (kept in s_deter); emit deter_n; stage bf16 A
    for (int i = tid; i < MB * DETER; i += THREADS) {
      int r = i >> 9, c = i & 511;
      const float* p = s_gemm + r * GRU3;
      float rr = sigmoidf_(p[c]);
      float cc = tanhf(rr * p[512 + c]);
      float uu = sigmoidf_(p[1024 + c] - 1.f);
      float dn = uu * cc + (1.f - uu) * s_deter[i];
      s_deter[i] = dn;
      out[(((size_t)(b0 + r)) * Tlen + t) * 704 + 192 + c] = dn;
      s_cat[r * 512 + c] = (bf16)dn;
    }
    __syncthreads();
    // ---- GEMM3: deter_n @ w_out (N=512, K=512)
    wmma_gemm(wT_out, s_cat, s_gemm, HIDDEN, HIDDEN, DETER, wave, lane);
    __syncthreads();
    // ---- LN(out)+silu -> x bf16
    ln_rows(s_gemm, HIDDEN, s_mv, tid);
    for (int i = tid; i < MB * HIDDEN; i += THREADS) {
      int r = i >> 9, c = i & 511;
      float y = (s_gemm[i] - s_mv[r * 2]) * s_mv[r * 2 + 1] * ln_out_g[c] + ln_out_b[c];
      y = y * sigmoidf_(y);
      s_cat[r * 512 + c] = (bf16)y;
    }
    __syncthreads();
    // ---- GEMM4: s = x @ w_ims (N=64, K=512)
    wmma_gemm(wT_ims, s_cat, s_stat, 64, 64, HIDDEN, wave, lane);
    __syncthreads();
    // ---- prior stats -> out; build cat3 = [deter_n, embed] (K=1536)
    for (int i = tid; i < MB * STOCH; i += THREADS) {
      int r = i >> 5, c = i & 31;
      float pm = s_stat[r * 64 + c] + b_ims[c];
      float ps = softplusf_(s_stat[r * 64 + 32 + c] + b_ims[32 + c]) + 0.1f;
      float ep = eps_prior[((size_t)(b0 + r) * Tlen + t) * STOCH + c];
      size_t ob = (((size_t)(b0 + r)) * Tlen + t) * 704;
      out[ob + 96 + c] = pm + ps * ep; out[ob + 128 + c] = pm; out[ob + 160 + c] = ps;
    }
    for (int i = tid; i < MB * GRU3; i += THREADS) {
      int r = i / GRU3, c = i - r * GRU3;
      float v = (c < DETER) ? s_deter[r * DETER + c]
                            : embed[((size_t)(b0 + r) * Tlen + t) * EMBED + (c - DETER)];
      s_cat[i] = (bf16)v;
    }
    __syncthreads();
    // ---- GEMM5: cat3 @ w_obs (N=512, K=1536)
    wmma_gemm(wT_obs, s_cat, s_gemm, HIDDEN, HIDDEN, GRU3, wave, lane);
    __syncthreads();
    // ---- LN(obs)+silu -> xo bf16
    ln_rows(s_gemm, HIDDEN, s_mv, tid);
    for (int i = tid; i < MB * HIDDEN; i += THREADS) {
      int r = i >> 9, c = i & 511;
      float y = (s_gemm[i] - s_mv[r * 2]) * s_mv[r * 2 + 1] * ln_obs_g[c] + ln_obs_b[c];
      y = y * sigmoidf_(y);
      s_cat[r * 512 + c] = (bf16)y;
    }
    __syncthreads();
    // ---- GEMM6: so = xo @ w_ostat (N=64, K=512)
    wmma_gemm(wT_ostat, s_cat, s_stat, 64, 64, HIDDEN, wave, lane);
    __syncthreads();
    // ---- posterior stats -> out; stoch <- ostoch
    for (int i = tid; i < MB * STOCH; i += THREADS) {
      int r = i >> 5, c = i & 31;
      float om = s_stat[r * 64 + c] + b_ostat[c];
      float os = softplusf_(s_stat[r * 64 + 32 + c] + b_ostat[32 + c]) + 0.1f;
      float eo = eps_post[((size_t)(b0 + r) * Tlen + t) * STOCH + c];
      float ost = om + os * eo;
      size_t ob = (((size_t)(b0 + r)) * Tlen + t) * 704;
      out[ob + c] = ost; out[ob + 32 + c] = om; out[ob + 64 + c] = os;
      s_stoch[r * STOCH + c] = ost;
    }
    __syncthreads();
  }
}

// ------------------------------------------------------------------- launch
extern "C" void kernel_launch(void* const* d_in, const int* in_sizes, int n_in,
                              void* d_out, int out_size, void* d_ws, size_t ws_size,
                              hipStream_t stream) {
  const float* embed    = (const float*)d_in[0];
  const float* action   = (const float*)d_in[1];
  const float* is_first = (const float*)d_in[2];
  const float* eps_pr   = (const float*)d_in[3];
  const float* eps_po   = (const float*)d_in[4];
  const float* W_init   = (const float*)d_in[5];
  const float* w_in     = (const float*)d_in[6];
  const float* ln_in_g  = (const float*)d_in[7];
  const float* ln_in_b  = (const float*)d_in[8];
  const float* w_gru    = (const float*)d_in[9];
  const float* ln_gru_g = (const float*)d_in[10];
  const float* ln_gru_b = (const float*)d_in[11];
  const float* w_out    = (const float*)d_in[12];
  const float* ln_out_g = (const float*)d_in[13];
  const float* ln_out_b = (const float*)d_in[14];
  const float* w_ims    = (const float*)d_in[15];
  const float* b_ims    = (const float*)d_in[16];
  const float* w_obs    = (const float*)d_in[17];
  const float* ln_obs_g = (const float*)d_in[18];
  const float* ln_obs_b = (const float*)d_in[19];
  const float* w_ostat  = (const float*)d_in[20];
  const float* b_ostat  = (const float*)d_in[21];

  char* ws = (char*)d_ws;
  bf16* wT_in    = (bf16*)(ws + OFF_WIN);
  bf16* wT_gru   = (bf16*)(ws + OFF_WGRU);
  bf16* wT_out   = (bf16*)(ws + OFF_WOUT);
  bf16* wT_ims   = (bf16*)(ws + OFF_WIMS);
  bf16* wT_obs   = (bf16*)(ws + OFF_WOBS);
  bf16* wT_ostat = (bf16*)(ws + OFF_WOSTAT);
  float* deter0  = (float*)(ws + OFF_DETER0);
  float* stoch0  = (float*)(ws + OFF_STOCH0);

  auto blocks = [](int n) { return (n + 255) / 256; };
  k_wt<<<blocks(512 * 64),    256, 0, stream>>>(w_in,    wT_in,    44,   512,  64);
  k_wt<<<blocks(1536 * 1024), 256, 0, stream>>>(w_gru,   wT_gru,   1024, 1536, 1024);
  k_wt<<<blocks(512 * 512),   256, 0, stream>>>(w_out,   wT_out,   512,  512,  512);
  k_wt<<<blocks(64 * 512),    256, 0, stream>>>(w_ims,   wT_ims,   512,  64,   512);
  k_wt<<<blocks(512 * 1536),  256, 0, stream>>>(w_obs,   wT_obs,   1536, 512,  1536);
  k_wt<<<blocks(64 * 512),    256, 0, stream>>>(w_ostat, wT_ostat, 512,  64,   512);
  k_init<<<1, 256, 0, stream>>>(W_init, w_out, ln_out_g, ln_out_b, w_ims, b_ims,
                                deter0, stoch0);
  k_rssm<<<NWG, THREADS, 0, stream>>>(embed, action, is_first, eps_pr, eps_po,
                                      ln_in_g, ln_in_b, ln_gru_g, ln_gru_b,
                                      ln_out_g, ln_out_b, b_ims, ln_obs_g, ln_obs_b,
                                      b_ostat, wT_in, wT_gru, wT_out, wT_ims, wT_obs,
                                      wT_ostat, deter0, stoch0, (float*)d_out);
}
